// RL3DFBackbone_knngate_34969623724725
// MI455X (gfx1250) — matrix operand order repaired and checked by reading, not compiled, not run on hardware
//
#include <hip/hip_runtime.h>
#include <hip/hip_bf16.h>

// ---------------------------------------------------------------------------
// CDNA5 / gfx1250 implementation. Heavy GEMMs (sparse 27-tap convs + BEV
// implicit-GEMM convs) run on v_wmma_f32_16x16x32_bf16 (fp32 accumulate),
// software-pipelined with double-buffered fragments so loads for tap k+1
// overlap the WMMAs of tap k (removes the wait-before-wmma serialization and
// the WMMA->VALU WAR hazard v_nops). Activations keep a bf16 mirror so WMMA
// A-fragments load straight from memory (global_load_b128 x2 per fragment).
// Attention is algebraically refactored (softmax sums to 1) to avoid the
// per-point 64x64x64 GEMM. Additive biases before BN cancel and are skipped.
// ---------------------------------------------------------------------------

typedef __attribute__((ext_vector_type(16))) __bf16 v16bf;
typedef __attribute__((ext_vector_type(8)))  float  v8f;

union FragU {
    unsigned short u[16];
    uint4          q[2];
    v16bf          v;
};

__device__ __forceinline__ unsigned short f2bf(float f) {
    __bf16 h = (__bf16)f;                       // RNE, lowers to v_cvt_*bf16_f32
    return __builtin_bit_cast(unsigned short, h);
}

// K(e) fragment map (16-bit operands, wave32):
//   lane: M/N = lane & 15, half = lane >> 4
//   element e (0..15): K = (e/8)*16 + half*8 + (e%8)
// => two contiguous runs of 8 values at offsets half*8 and 16+half*8.

// ------------------------------ weight packing -----------------------------

// Wconv (12, 27, 64, 64) fp32 -> pw blocks: blk = ((lbrj*27+k)*8 + dT*2 + chunk)
__global__ void k_pack_wconv(const float* __restrict__ W, unsigned short* __restrict__ pw) {
    int id   = blockIdx.x * 256 + threadIdx.x;          // < 1,327,104
    int e    = id & 15;
    int lane = (id >> 4) & 31;
    int blk  = id >> 9;
    int chunk = blk & 1;
    int dT    = (blk >> 1) & 3;
    int k     = (blk >> 3) % 27;
    int lbrj  = blk / 216;                               // 8*27
    int c = chunk * 32 + (e >> 3) * 16 + (lane >> 4) * 8 + (e & 7);
    int d = dT * 16 + (lane & 15);
    pw[id] = f2bf(W[(((size_t)(lbrj * 27 + k)) * 64 + c) * 64 + d]);
}

// Wbev (4, 1024, 64, 3, 3) fp32 -> pw blocks: blk = ((lb*288 + t)*4 + dT),
// t = cell*32 + icc. Kernel-flip folded in: offset (dy-1,dx-1) -> taps (2-dy,2-dx).
__global__ void k_pack_wbev(const float* __restrict__ W, unsigned short* __restrict__ pw) {
    int id   = blockIdx.x * 256 + threadIdx.x;          // < 2,359,296
    int e    = id & 15;
    int lane = (id >> 4) & 31;
    int blk  = id >> 9;
    int dT   = blk & 3;
    int icc  = (blk >> 2) & 31;
    int cell = (blk >> 7) % 9;
    int lb   = blk / 1152;                               // 4*32*9
    int ic = icc * 32 + (e >> 3) * 16 + (lane >> 4) * 8 + (e & 7);
    int oc = dT * 16 + (lane & 15);
    int dy = cell / 3, dx = cell % 3;
    int ky = 2 - dy,  kx = 2 - dx;
    pw[id] = f2bf(W[((((size_t)lb * 1024 + ic) * 64 + oc) * 3 + ky) * 3 + kx]);
}

// ------------------------------ input linear (fp32 + bf16 mirror) ----------

__global__ void k_linear(const float* __restrict__ feat, const float* __restrict__ Wp,
                         const float* __restrict__ bp, float* __restrict__ f,
                         unsigned short* __restrict__ fbf, int n) {
    int id = blockIdx.x * 256 + threadIdx.x;
    if (id >= n * 64) return;
    int v = id >> 6, c = id & 63;
    float a = bp[c];
#pragma unroll
    for (int i = 0; i < 4; ++i) a += feat[v * 4 + i] * Wp[i * 64 + c];
    f[id]   = a;
    fbf[id] = f2bf(a);
}

// ------------------------------ fragment load helpers ----------------------

__device__ __forceinline__ void loadFrag(const unsigned short* p, FragU& F) {
    F.q[0] = ((const uint4*)p)[0];
    F.q[1] = ((const uint4*)p)[1];
}

__device__ __forceinline__ void loadFragA(const unsigned short* p, FragU& F) {
    F.q[0] = ((const uint4*)p)[0];
    F.q[1] = ((const uint4*)(p + 16))[0];
}

__device__ __forceinline__ void loadA_sconv(const unsigned short* __restrict__ fbf,
                                            const int* __restrict__ nbrRow, int k,
                                            int half, FragU& A0, FragU& A1) {
    int idx = nbrRow[k];
    if (idx >= 0) {
        const unsigned short* fp = fbf + (size_t)idx * 64 + half * 8;
        A0.q[0] = *(const uint4*)(fp);
        A0.q[1] = *(const uint4*)(fp + 16);
        A1.q[0] = *(const uint4*)(fp + 32);
        A1.q[1] = *(const uint4*)(fp + 48);
    } else {
        A0.q[0] = make_uint4(0, 0, 0, 0); A0.q[1] = make_uint4(0, 0, 0, 0);
        A1.q[0] = make_uint4(0, 0, 0, 0); A1.q[1] = make_uint4(0, 0, 0, 0);
    }
}

// ------------------------------ sparse conv (WMMA, pipelined) --------------
// Tile: 16 voxels (M) x 64 out-ch (4 waves x 16 N). 27 taps x 2 K-chunks of 32.

__global__ __launch_bounds__(128) void k_sconv(const unsigned short* __restrict__ fbf,
                                               const int* __restrict__ nbr,
                                               const unsigned short* __restrict__ pw,
                                               float* __restrict__ tmp,
                                               float* __restrict__ stats,
                                               int lbrj) {
    const int lane = threadIdx.x & 31;
    const int w    = threadIdx.x >> 5;        // N-tile
    const int M    = lane & 15;
    const int half = lane >> 4;
    const int n0   = blockIdx.x * 16;
    const int* nbrRow = nbr + (size_t)(n0 + M) * 27;

    v8f acc = {0.f, 0.f, 0.f, 0.f, 0.f, 0.f, 0.f, 0.f};

    const unsigned short* pk = pw + ((size_t)(lbrj * 27) * 8 + w * 2) * 512 + lane * 16;

    FragU aA0, aA1, bA0, bA1;                  // buffer A (even taps)
    FragU aB0, aB1, bB0, bB1;                  // buffer B (odd taps)

    // prologue: tap 0 -> buffer A
    loadA_sconv(fbf, nbrRow, 0, half, aA0, aA1);
    loadFrag(pk, bA0); loadFrag(pk + 512, bA1);

    for (int k = 0; k < 26; k += 2) {
        // issue tap k+1 loads, then consume tap k
        loadA_sconv(fbf, nbrRow, k + 1, half, aB0, aB1);
        loadFrag(pk + 4096, bB0); loadFrag(pk + 4096 + 512, bB1);
        acc = __builtin_amdgcn_wmma_f32_16x16x32_bf16(false, aA0.v, false, bA0.v, (short)0, acc, false, false);
        acc = __builtin_amdgcn_wmma_f32_16x16x32_bf16(false, aA1.v, false, bA1.v, (short)0, acc, false, false);

        // issue tap k+2 loads, then consume tap k+1
        loadA_sconv(fbf, nbrRow, k + 2, half, aA0, aA1);
        loadFrag(pk + 8192, bA0); loadFrag(pk + 8192 + 512, bA1);
        acc = __builtin_amdgcn_wmma_f32_16x16x32_bf16(false, aB0.v, false, bB0.v, (short)0, acc, false, false);
        acc = __builtin_amdgcn_wmma_f32_16x16x32_bf16(false, aB1.v, false, bB1.v, (short)0, acc, false, false);

        pk += 8192;
    }
    // epilogue: tap 26 sits in buffer A
    acc = __builtin_amdgcn_wmma_f32_16x16x32_bf16(false, aA0.v, false, bA0.v, (short)0, acc, false, false);
    acc = __builtin_amdgcn_wmma_f32_16x16x32_bf16(false, aA1.v, false, bA1.v, (short)0, acc, false, false);

    int d = w * 16 + (lane & 15);
    float s = 0.f, s2 = 0.f;
#pragma unroll
    for (int r = 0; r < 8; ++r) {
        int row = n0 + half * 8 + r;
        float v = acc[r];
        tmp[(size_t)row * 64 + d] = v;
        s += v; s2 += v * v;
    }
    atomicAdd(stats + d, s);
    atomicAdd(stats + 64 + d, s2);
}

// ------------------------------ BN + ReLU (per-channel over n) -------------

__global__ void k_bnrelu(const float* __restrict__ tmp, const float* __restrict__ stats,
                         const float* __restrict__ g, const float* __restrict__ b,
                         float* __restrict__ f, unsigned short* __restrict__ fbf,
                         int n, float invCnt) {
    int id = blockIdx.x * 256 + threadIdx.x;
    if (id >= n * 64) return;
    int c = id & 63;
    float m = stats[c] * invCnt;
    float v = stats[64 + c] * invCnt - m * m;
    float y = (tmp[id] - m) * rsqrtf(v + 1e-5f) * g[c] + b[c];
    y = fmaxf(y, 0.f);
    f[id]   = y;
    fbf[id] = f2bf(y);
}

// ------------------------------ image gate ---------------------------------

__global__ void k_imgf(const float* __restrict__ img, const float* __restrict__ Wimg,
                       const float* __restrict__ bimg, float* __restrict__ out) {
    int t = threadIdx.x;                 // 128 threads: (b, c)
    int b = t >> 6, c = t & 63;
    float a = bimg[c];
    for (int i = 0; i < 512; ++i) a += img[b * 512 + i] * Wimg[i * 64 + c];
    out[t] = 1.f / (1.f + __expf(-a));
}

// ------------------------------ KNN attention + gate -----------------------
// fused = (softmax(fL.key) @ key) @ Wval + bval   (softmax sums to 1)

__global__ __launch_bounds__(64) void k_attn(float* __restrict__ fL, unsigned short* __restrict__ fLbf,
                                             const float* __restrict__ fR,
                                             const int* __restrict__ knn,
                                             const float* __restrict__ Wval, const float* __restrict__ bval,
                                             const float* __restrict__ Wgate, const float* __restrict__ bgate,
                                             const float* __restrict__ imgf, const int* __restrict__ coordsL,
                                             int K) {
    __shared__ float fLs[64], sc[64], pooled[64], fused[64];
    __shared__ int   ki[64];
    __shared__ float red;
    int n = blockIdx.x, t = threadIdx.x;

    fLs[t] = fL[(size_t)n * 64 + t];
    __syncthreads();

    if (t < K) {
        int idx = knn[(size_t)n * K + t];
        ki[t] = idx;
        const float* kp = fR + (size_t)idx * 64;
        float a = 0.f;
        for (int c = 0; c < 64; ++c) a += fLs[c] * kp[c];
        sc[t] = a;
    }
    __syncthreads();

    if (t == 0) {
        float mx = -1e30f;
        for (int k = 0; k < K; ++k) mx = fmaxf(mx, sc[k]);
        float s = 0.f;
        for (int k = 0; k < K; ++k) { float e = __expf(sc[k] - mx); sc[k] = e; s += e; }
        red = 1.f / s;
    }
    __syncthreads();
    float inv = red;

    {   // pooled[c] = sum_k attn_k * key_k[c]
        float a = 0.f;
        for (int k = 0; k < K; ++k) a += sc[k] * fR[(size_t)ki[k] * 64 + t];
        pooled[t] = a * inv;
    }
    __syncthreads();

    int b = coordsL[n * 4];
    {
        float a = bval[t];
        for (int e = 0; e < 64; ++e) a += pooled[e] * Wval[e * 64 + t];
        fused[t] = a * imgf[b * 64 + t];
    }
    __syncthreads();

    {
        float a = bgate[t];
        for (int e = 0; e < 64; ++e) a += fLs[e]   * Wgate[e * 64 + t];
        for (int e = 0; e < 64; ++e) a += fused[e] * Wgate[(64 + e) * 64 + t];
        float g = 1.f / (1.f + __expf(-a));
        float y = fLs[t] + g * fused[t];
        fL[(size_t)n * 64 + t]   = y;
        fLbf[(size_t)n * 64 + t] = f2bf(y);
    }
}

// ------------------------------ scatter to padded dense BEV (bf16) ---------
// dense layout (B, 130, 130, 1024), channel ic = c*16 + z. Pure bf16 copy.

__global__ void k_scatter(const unsigned short* __restrict__ fbf, const int* __restrict__ coords,
                          unsigned short* __restrict__ dense, int n) {
    int id = blockIdx.x * 256 + threadIdx.x;
    if (id >= n * 64) return;
    int v = id >> 6, c = id & 63;
    int b = coords[v * 4 + 0], z = coords[v * 4 + 1];
    int y = coords[v * 4 + 2], x = coords[v * 4 + 3];
    dense[((size_t)(b * 130 + y + 1) * 130 + (x + 1)) * 1024 + c * 16 + z] =
        fbf[(size_t)v * 64 + c];
}

// ------------------------------ BEV conv (implicit GEMM, WMMA, pipelined) --
// Tile: 16 pixels along x (M) x 64 oc (4 waves x 16 N);
// K flattened: t = cell*32 + icc, 288 chunks of 32.

__device__ __forceinline__ const unsigned short* bevA_addr(
    const unsigned short* __restrict__ dense, int b, int y, int xBM, int half, int t) {
    int cell = t >> 5, icc = t & 31;
    int dy = cell / 3, dx = cell % 3;
    return dense + ((size_t)(b * 130 + y + dy) * 130 + (xBM + dx)) * 1024 + half * 8 + icc * 32;
}

__global__ __launch_bounds__(128) void k_bevconv(const unsigned short* __restrict__ dense,
                                                 const unsigned short* __restrict__ pw,
                                                 float* __restrict__ bevTmp,
                                                 float* __restrict__ stats,
                                                 int lb) {
    const int lane = threadIdx.x & 31;
    const int w    = threadIdx.x >> 5;
    const int M    = lane & 15;
    const int half = lane >> 4;
    const int bid  = blockIdx.x;                 // 2048 = B * 128 * 8
    const int xB   = (bid & 7) * 16;
    const int y    = (bid >> 3) & 127;
    const int b    = bid >> 10;
    const int xBM  = xB + M;

    const unsigned short* pwBase =
        pw + (size_t)lb * 1152 * 512 + (size_t)w * 512 + lane * 16;   // + t*2048

    v8f acc = {0.f, 0.f, 0.f, 0.f, 0.f, 0.f, 0.f, 0.f};

    FragU actA, wgtA, actB, wgtB;

    __builtin_prefetch(pwBase, 0, 1);
    loadFragA(bevA_addr(dense, b, y, xBM, half, 0), actA);
    loadFrag(pwBase, wgtA);

    for (int t = 0; t < 286; t += 2) {
        loadFragA(bevA_addr(dense, b, y, xBM, half, t + 1), actB);
        loadFrag(pwBase + (size_t)(t + 1) * 2048, wgtB);
        acc = __builtin_amdgcn_wmma_f32_16x16x32_bf16(false, actA.v, false, wgtA.v, (short)0, acc, false, false);

        loadFragA(bevA_addr(dense, b, y, xBM, half, t + 2), actA);
        loadFrag(pwBase + (size_t)(t + 2) * 2048, wgtA);
        acc = __builtin_amdgcn_wmma_f32_16x16x32_bf16(false, actB.v, false, wgtB.v, (short)0, acc, false, false);
    }
    // t=286 sits in A buffers; load final t=287 then drain both
    loadFragA(bevA_addr(dense, b, y, xBM, half, 287), actB);
    loadFrag(pwBase + (size_t)287 * 2048, wgtB);
    acc = __builtin_amdgcn_wmma_f32_16x16x32_bf16(false, actA.v, false, wgtA.v, (short)0, acc, false, false);
    acc = __builtin_amdgcn_wmma_f32_16x16x32_bf16(false, actB.v, false, wgtB.v, (short)0, acc, false, false);

    int oc = w * 16 + M;
    float s = 0.f, s2 = 0.f;
#pragma unroll
    for (int r = 0; r < 8; ++r) {
        int x = xB + half * 8 + r;
        float v = acc[r];
        bevTmp[((size_t)(b * 64 + oc) * 128 + y) * 128 + x] = v;
        s += v; s2 += v * v;
    }
    atomicAdd(stats + oc, s);
    atomicAdd(stats + 64 + oc, s2);
}

// ------------------------------ BEV BN + ReLU -> output --------------------

__global__ void k_bevbn(const float* __restrict__ t, const float* __restrict__ stats,
                        const float* __restrict__ g, const float* __restrict__ bb,
                        float* __restrict__ out, int lb) {
    int id = blockIdx.x * 256 + threadIdx.x;       // 2*64*128*128
    int x  = id & 127;
    int y  = (id >> 7) & 127;
    int oc = (id >> 14) & 63;
    int b  = id >> 20;
    float m = stats[oc] * (1.f / 32768.f);
    float v = stats[64 + oc] * (1.f / 32768.f) - m * m;
    float val = (t[id] - m) * rsqrtf(v + 1e-5f) * g[oc] + bb[oc];
    out[((size_t)(b * 256 + lb * 64 + oc) * 128 + y) * 128 + x] = fmaxf(val, 0.f);
}

// ---------------------------------------------------------------------------

extern "C" void kernel_launch(void* const* d_in, const int* in_sizes, int n_in,
                              void* d_out, int out_size, void* d_ws, size_t ws_size,
                              hipStream_t stream) {
    (void)in_sizes; (void)n_in; (void)out_size; (void)ws_size;

    const float* featR  = (const float*)d_in[0];
    const float* featL  = (const float*)d_in[1];
    const float* img    = (const float*)d_in[2];
    const float* Win    = (const float*)d_in[3];
    const float* bin_   = (const float*)d_in[4];
    const float* Wconv  = (const float*)d_in[5];
    // d_in[6] bconv, d_in[16] bbev: additive bias before BN cancels -> skipped
    const float* bng    = (const float*)d_in[7];
    const float* bnb    = (const float*)d_in[8];
    const float* Wimg   = (const float*)d_in[9];
    const float* bimg   = (const float*)d_in[10];
    const float* Wval   = (const float*)d_in[11];
    const float* bval   = (const float*)d_in[12];
    const float* Wgate  = (const float*)d_in[13];
    const float* bgate  = (const float*)d_in[14];
    const float* Wbev   = (const float*)d_in[15];
    const float* bntg   = (const float*)d_in[17];
    const float* bntb   = (const float*)d_in[18];
    const int* nbrR     = (const int*)d_in[19];
    const int* nbrL     = (const int*)d_in[20];
    const int* knn0     = (const int*)d_in[21];
    const int* knn1     = (const int*)d_in[22];
    const int* coordsR  = (const int*)d_in[23];
    const int* coordsL  = (const int*)d_in[24];

    const int NR = 24000, NL = 40000;

    char* ws = (char*)d_ws;
    size_t off = 0;
    auto alloc = [&](size_t bytes) -> void* {
        void* p = ws + off;
        off = (off + bytes + 255) & ~(size_t)255;
        return p;
    };
    float* fR              = (float*)alloc((size_t)NR * 64 * 4);
    float* fL              = (float*)alloc((size_t)NL * 64 * 4);
    unsigned short* fbfR   = (unsigned short*)alloc((size_t)NR * 64 * 2);
    unsigned short* fbfL   = (unsigned short*)alloc((size_t)NL * 64 * 2);
    float* tmp             = (float*)alloc((size_t)NL * 64 * 4);
    unsigned short* pwConv = (unsigned short*)alloc(1327104ull * 2);
    unsigned short* pwBev  = (unsigned short*)alloc(2359296ull * 2);
    unsigned short* dense  = (unsigned short*)alloc((size_t)2 * 130 * 130 * 1024 * 2);
    float* bevTmp          = (float*)alloc((size_t)2 * 64 * 128 * 128 * 4);
    float* stats           = (float*)alloc(128 * 4);
    float* imgfbuf         = (float*)alloc(128 * 4);

    k_pack_wconv<<<1327104 / 256, 256, 0, stream>>>(Wconv, pwConv);
    k_pack_wbev <<<2359296 / 256, 256, 0, stream>>>(Wbev, pwBev);

    k_linear<<<(NR * 64) / 256, 256, 0, stream>>>(featR, Win,        bin_,      fR, fbfR, NR);
    k_linear<<<(NL * 64) / 256, 256, 0, stream>>>(featL, Win + 256,  bin_ + 64, fL, fbfL, NL);

    for (int l = 0; l < 2; ++l) {
        for (int br = 0; br < 2; ++br) {
            float* f            = br ? fL   : fR;
            unsigned short* fbf = br ? fbfL : fbfR;
            const int* nbr      = br ? nbrL : nbrR;
            int N               = br ? NL   : NR;
            for (int j = 0; j < 3; ++j) {
                int lbrj = (l * 2 + br) * 3 + j;
                hipMemsetAsync(stats, 0, 128 * 4, stream);
                k_sconv<<<N / 16, 128, 0, stream>>>(fbf, nbr, pwConv, tmp, stats, lbrj);
                k_bnrelu<<<(N * 64) / 256, 256, 0, stream>>>(tmp, stats, bng + lbrj * 64,
                                                             bnb + lbrj * 64, f, fbf, N, 1.0f / N);
            }
        }

        k_imgf<<<1, 128, 0, stream>>>(img, Wimg + l * 512 * 64, bimg + l * 64, imgfbuf);
        const int* knn = l ? knn1 : knn0;
        int K          = l ? 32 : 64;
        k_attn<<<NL, 64, 0, stream>>>(fL, fbfL, fR, knn, Wval + l * 4096, bval + l * 64,
                                      Wgate + l * 8192, bgate + l * 64, imgfbuf, coordsL, K);

        for (int br = 0; br < 2; ++br) {
            int lb              = l * 2 + br;
            unsigned short* fbf = br ? fbfL    : fbfR;
            const int* coords   = br ? coordsL : coordsR;
            int N               = br ? NL      : NR;
            hipMemsetAsync(dense, 0, (size_t)2 * 130 * 130 * 1024 * 2, stream);
            k_scatter<<<(N * 64) / 256, 256, 0, stream>>>(fbf, coords, dense, N);
            hipMemsetAsync(stats, 0, 128 * 4, stream);
            k_bevconv<<<2048, 128, 0, stream>>>(dense, pwBev, bevTmp, stats, lb);
            k_bevbn<<<(2 * 64 * 128 * 128) / 256, 256, 0, stream>>>(bevTmp, stats, bntg + lb * 64,
                                                                    bntb + lb * 64, (float*)d_out, lb);
        }
    }
}
